// AttPABDecoder_35072702939371
// MI455X (gfx1250) — matrix operand order repaired and checked by reading, not compiled, not run on hardware
//
#include <hip/hip_runtime.h>
#include <hip/hip_bf16.h>
#include <stddef.h>

// ---------------------------------------------------------------------------
// AttPABDecoder for MI455X (gfx1250).
// Key idea: the per-step cost is dominated by the V=50000 vocab head
// (2 x [16x512]x[512x50000]). In fp32 the weights are 204.8MB/step and miss
// the 192MB L2; cast once to bf16 (102.4MB) so all 32 serial steps run out
// of L2, and do the head with v_wmma_f32_16x16x32_bf16 (M=16 == WMMA tile).
// Workspace assumption: ws_size >= ~110MB (2x 51.2MB bf16 weights + ~6MB).
// ---------------------------------------------------------------------------

#define B_   16
#define E_   512
#define H_   512
#define DV_  512
#define V_   50000
#define NROWS_ 64
#define NTAGS_ 8
#define TAGLEN_ 8
#define STEPS_ 32
#define BT_  (B_ * NTAGS_)     // 128

typedef __bf16 v16bf __attribute__((ext_vector_type(16)));
typedef __bf16 v8bf  __attribute__((ext_vector_type(8)));
typedef float  v8f   __attribute__((ext_vector_type(8)));

__device__ __forceinline__ float sigmoidf_(float x) {
    return 1.0f / (1.0f + __expf(-x));
}

// ---------------------------------------------------------------------------
// Setup kernels
// ---------------------------------------------------------------------------
__global__ void cvt_bf16_kernel(const float* __restrict__ in,
                                __bf16* __restrict__ out, int n) {
    int i = blockIdx.x * blockDim.x + threadIdx.x;
    int stride = gridDim.x * blockDim.x;
    for (; i < n; i += stride) out[i] = (__bf16)in[i];
}

__global__ void init_kernel(const float* __restrict__ hidden_in,
                            float* __restrict__ h_cur,
                            float* __restrict__ tags_h,
                            int* __restrict__ ntok) {
    int i = blockIdx.x * blockDim.x + threadIdx.x;
    if (i < 2 * B_ * H_) h_cur[i] = hidden_in[i];
    if (i < BT_ * E_)    tags_h[i] = 0.0f;
    if (i < B_)          ntok[i] = 1;   // START token
}

__global__ void embed_lg_kernel(const int* __restrict__ loc,
                                const int* __restrict__ gender,
                                const float* __restrict__ loc_emb,
                                const float* __restrict__ gen_emb,
                                float* __restrict__ loc_e,
                                float* __restrict__ gen_e) {
    int i = blockIdx.x * blockDim.x + threadIdx.x;
    if (i >= B_ * E_) return;
    int b = i >> 9, d = i & 511;
    loc_e[i] = loc_emb[(size_t)loc[b] * E_ + d];
    gen_e[i] = gen_emb[(size_t)gender[b] * E_ + d];
}

__global__ void tag_gather_kernel(const float* __restrict__ text_emb,
                                  const int* __restrict__ tags_token,
                                  int t, float* __restrict__ xt) {
    int i = blockIdx.x * blockDim.x + threadIdx.x;
    if (i >= BT_ * E_) return;
    int seq = i >> 9, d = i & 511;
    int token = tags_token[seq * TAGLEN_ + t];
    xt[i] = text_emb[(size_t)token * E_ + d];
}

__global__ void tag_select_kernel(const float* __restrict__ hs_all,
                                  const int* __restrict__ tags_length,
                                  float* __restrict__ enc_tags) {
    int i = blockIdx.x * blockDim.x + threadIdx.x;
    if (i >= BT_ * E_) return;
    int seq = i >> 9;
    int t = tags_length[seq] - 1;
    t = (t < 0) ? 0 : (t > TAGLEN_ - 1 ? TAGLEN_ - 1 : t);
    enc_tags[i] = hs_all[(size_t)t * BT_ * E_ + i];
}

// ---------------------------------------------------------------------------
// Generic single-step GRU layer (PyTorch gate order r,z,n).
// One thread per (row, hidden-dim). Used for both the tag encoder (nrows=128)
// and the decoder layers (nrows=16).
// ---------------------------------------------------------------------------
__global__ void gru_layer_kernel(const float* __restrict__ x, int xdim,
                                 const float* __restrict__ h_in,
                                 const float* __restrict__ W_ih,
                                 const float* __restrict__ W_hh,
                                 const float* __restrict__ b_ih,
                                 const float* __restrict__ b_hh,
                                 float* __restrict__ h_out,
                                 __bf16* __restrict__ state_bf,   // nullable
                                 float* __restrict__ hs_copy,     // nullable
                                 int nrows) {
    int idx = blockIdx.x * blockDim.x + threadIdx.x;
    if (idx >= nrows * 512) return;
    int row = idx >> 9, d = idx & 511;
    const float* xb = x + (size_t)row * xdim;
    const float* hb = h_in + (size_t)row * 512;
    float gi[3], gh[3];
#pragma unroll
    for (int g = 0; g < 3; ++g) {
        const float* wi = W_ih + (size_t)(g * 512 + d) * xdim;
        float ai = b_ih[g * 512 + d];
        for (int k = 0; k < xdim; ++k) ai = __fmaf_rn(xb[k], wi[k], ai);
        gi[g] = ai;
        const float* wh = W_hh + (size_t)(g * 512 + d) * 512;
        float ah = b_hh[g * 512 + d];
        for (int k = 0; k < 512; ++k) ah = __fmaf_rn(hb[k], wh[k], ah);
        gh[g] = ah;
    }
    float r = sigmoidf_(gi[0] + gh[0]);
    float z = sigmoidf_(gi[1] + gh[1]);
    float n = tanhf(gi[2] + r * gh[2]);
    float hn = (1.0f - z) * n + z * hb[d];
    h_out[idx] = hn;
    if (state_bf) state_bf[idx] = (__bf16)hn;
    if (hs_copy)  hs_copy[idx] = hn;
}

// ---------------------------------------------------------------------------
// Attention stage helper: score = softmax(mask((s_h W) . v_n)), out = score.v
// Block-wide (256 threads). All pointers generic (may be LDS or global).
// ---------------------------------------------------------------------------
__device__ void attend_stage(const float* __restrict__ W,
                             const float* sh, float* hw, float* red, float* sc,
                             const float* vbase, int vstride,
                             const int* mask, int nvals,
                             float* outv, int tid) {
    // hw = s_h @ W  (W is [512,512], column access coalesced across lanes)
    for (int d = tid; d < 512; d += 256) {
        float acc = 0.0f;
        for (int k = 0; k < 512; ++k) acc = __fmaf_rn(sh[k], W[k * 512 + d], acc);
        hw[d] = acc;
    }
    __syncthreads();
    // partial dot products: 4 partials per value row
    if (tid < nvals * 4) {
        int n = tid >> 2, part = tid & 3;
        const float* v = vbase + (size_t)n * vstride + part * 128;
        const float* h = hw + part * 128;
        float acc = 0.0f;
        for (int k = 0; k < 128; ++k) acc = __fmaf_rn(h[k], v[k], acc);
        red[tid] = acc;
    }
    __syncthreads();
    // softmax (serial over <=64 entries)
    if (tid == 0) {
        float mx = -1e30f;
        for (int n = 0; n < nvals; ++n) {
            float l = red[4 * n] + red[4 * n + 1] + red[4 * n + 2] + red[4 * n + 3];
            if (mask && mask[n] <= 0) l = -1e9f;
            sc[n] = l;
            mx = fmaxf(mx, l);
        }
        float s = 0.0f;
        for (int n = 0; n < nvals; ++n) { float e = __expf(sc[n] - mx); sc[n] = e; s += e; }
        float inv = 1.0f / s;
        for (int n = 0; n < nvals; ++n) sc[n] *= inv;
    }
    __syncthreads();
    // weighted sum of value rows
    for (int d = tid; d < 512; d += 256) {
        float acc = 0.0f;
        for (int n = 0; n < nvals; ++n)
            acc = __fmaf_rn(sc[n], vbase[(size_t)n * vstride + d], acc);
        outv[d] = acc;
    }
    __syncthreads();
}

// One block per batch row: embedding gather, 3 attentions, builds x=[emb|ctx]
// and the persona vector wp (stored bf16 for the WMMA head).
__global__ void attend_kernel(const float* __restrict__ h_top,      // [B,H] = h[-1]
                              const int* __restrict__ ntok,
                              const float* __restrict__ text_emb,
                              const float* __restrict__ W_dec,
                              const float* __restrict__ attn_value, // [B,64,DV]
                              const int* __restrict__ attn_mask,    // [B,64]
                              const float* __restrict__ W_tags,
                              const float* __restrict__ enc_tags,   // [B,8,E]
                              const int* __restrict__ tags_length,  // [B,8] (>0 == valid)
                              const float* __restrict__ W_prof,
                              const float* __restrict__ loc_e,      // [B,E]
                              const float* __restrict__ gen_e,      // [B,E]
                              float* __restrict__ xbuf,             // [B, E+DV]
                              __bf16* __restrict__ wp_bf) {         // [B,E]
    __shared__ float sh[512], hw[512], red[256], sc[64], vec[512], prof[3 * 512];
    int b = blockIdx.x, tid = threadIdx.x;
    for (int d = tid; d < 512; d += 256) sh[d] = h_top[b * 512 + d];
    int token = ntok[b];
    for (int d = tid; d < 512; d += 256)
        xbuf[(size_t)b * 1024 + d] = text_emb[(size_t)token * E_ + d];
    __syncthreads();

    // decoder attention -> ctx
    attend_stage(W_dec, sh, hw, red, sc,
                 attn_value + (size_t)b * NROWS_ * DV_, DV_,
                 attn_mask + b * NROWS_, NROWS_, vec, tid);
    for (int d = tid; d < 512; d += 256) xbuf[(size_t)b * 1024 + 512 + d] = vec[d];
    __syncthreads();

    // tags attention -> wt (mask>0 iff length>0)
    attend_stage(W_tags, sh, hw, red, sc,
                 enc_tags + (size_t)b * NTAGS_ * E_, E_,
                 tags_length + b * NTAGS_, NTAGS_, vec, tid);

    // profile = [loc_e, gen_e, wt]
    for (int d = tid; d < 512; d += 256) {
        prof[d]        = loc_e[b * 512 + d];
        prof[512 + d]  = gen_e[b * 512 + d];
        prof[1024 + d] = vec[d];
    }
    __syncthreads();
    attend_stage(W_prof, sh, hw, red, sc, prof, 512, nullptr, 3, vec, tid);
    for (int d = tid; d < 512; d += 256) wp_bf[b * 512 + d] = (__bf16)vec[d];
}

// ---------------------------------------------------------------------------
// Persona gate a = sigmoid(state @ w_o.T). 16 waves, one per batch row.
// ---------------------------------------------------------------------------
__global__ void gate_kernel(const float* __restrict__ state,
                            const float* __restrict__ w_o,
                            float* __restrict__ a) {
    int b = threadIdx.x >> 5;
    int lane = threadIdx.x & 31;
    float acc = 0.0f;
    for (int k = lane; k < 512; k += 32)
        acc = __fmaf_rn(state[b * 512 + k], w_o[k], acc);
    for (int off = 16; off > 0; off >>= 1) acc += __shfl_down(acc, off, 32);
    if (lane == 0) a[b] = sigmoidf_(acc);
}

// ---------------------------------------------------------------------------
// Vocab head: logits = (1-a)*(wp @ Wp^T) + a*(state @ Ws^T) + bias_o
// bf16 WMMA, one 16x16 N-tile per wave, K=512 in 16 chunks of 32.
// A layout (16-bit 16x32): lane<16 -> row=lane, K c..c+15 (contiguous);
//                          lane>=16 -> row=lane-16, K c+16..c+31 (contiguous).
// B layout (16-bit 32x16): lane<16 -> N=lane,    K c..c+15 (contiguous);
//                          lane>=16 -> N=lane-16, K c+16..c+31 (contiguous).
// Since Wp is [V,E] row-major and B = Wp^T, each fragment is one 32B load.
// ---------------------------------------------------------------------------
__global__ void logits_wmma_kernel(const __bf16* __restrict__ Ap,   // wp bf16 [16,512]
                                   const __bf16* __restrict__ As,   // state bf16 [16,512]
                                   const __bf16* __restrict__ Wp,   // [V,512] bf16
                                   const __bf16* __restrict__ Ws,   // [V,512] bf16
                                   const float* __restrict__ a,     // [16]
                                   const float* __restrict__ bias_o,// [V]
                                   float* __restrict__ out,         // [B,STEPS,V]
                                   int step) {
    const int NTILES = V_ / 16;                   // 3125 exactly
    int wave = threadIdx.x >> 5;
    int lane = threadIdx.x & 31;
    int tile = blockIdx.x * (blockDim.x >> 5) + wave;
    if (tile >= NTILES) return;                   // wave-uniform exit
    int n0 = tile * 16;
    int hi = lane >> 4;                           // 0 or 1
    int lm = lane & 15;
    int n  = n0 + lm;                             // output / B column
    int aoff = hi * 16;                           // A K-sub-offset per half-wave

    const __bf16* wpRow = Wp + (size_t)n * E_ + hi * 16;
    const __bf16* wsRow = Ws + (size_t)n * E_ + hi * 16;
    const __bf16* apRow = Ap + (size_t)lm * E_;
    const __bf16* asRow = As + (size_t)lm * E_;

    v8f cp = {};
    v8f cs = {};
#pragma unroll 4
    for (int kc = 0; kc < E_; kc += 32) {
        v16bf ap, as2, bp, bs;
        {
            const __bf16* base = apRow + kc + aoff;
            v8bf lo = *(const v8bf*)(base);
            v8bf hi8 = *(const v8bf*)(base + 8);
#pragma unroll
            for (int i = 0; i < 8; ++i) { ap[i] = lo[i]; ap[8 + i] = hi8[i]; }
        }
        {
            const __bf16* base = asRow + kc + aoff;
            v8bf lo = *(const v8bf*)(base);
            v8bf hi8 = *(const v8bf*)(base + 8);
#pragma unroll
            for (int i = 0; i < 8; ++i) { as2[i] = lo[i]; as2[8 + i] = hi8[i]; }
        }
        bp = *(const v16bf*)(wpRow + kc);
        bs = *(const v16bf*)(wsRow + kc);
        cp = __builtin_amdgcn_wmma_f32_16x16x32_bf16(false, ap, false, bp,
                                                     (short)0, cp, false, false);
        cs = __builtin_amdgcn_wmma_f32_16x16x32_bf16(false, as2, false, bs,
                                                     (short)0, cs, false, false);
    }

    float bias = bias_o[n];
    // D layout: VGPR r -> row = r + 8*hi (batch), col = lm
#pragma unroll
    for (int r = 0; r < 8; ++r) {
        int row = r + hi * 8;
        float av = a[row];
        float val = (1.0f - av) * cp[r] + av * cs[r] + bias;
        out[(size_t)row * STEPS_ * V_ + (size_t)step * V_ + n] = val;
    }
}

// ---------------------------------------------------------------------------
// argmax over this step's logits row (first-max-index semantics) -> next token
// ---------------------------------------------------------------------------
__global__ void argmax_kernel(const float* __restrict__ out, int step,
                              int* __restrict__ ntok) {
    int b = blockIdx.x;
    const float* row = out + (size_t)b * STEPS_ * V_ + (size_t)step * V_;
    float best = -3.4e38f;
    int bidx = 0;
    for (int v = threadIdx.x; v < V_; v += blockDim.x) {
        float val = row[v];
        if (val > best) { best = val; bidx = v; }   // strided: keeps lowest idx per thread
    }
    __shared__ float sv[256];
    __shared__ int   si[256];
    sv[threadIdx.x] = best;
    si[threadIdx.x] = bidx;
    __syncthreads();
    for (int s = 128; s > 0; s >>= 1) {
        if ((int)threadIdx.x < s) {
            float ov = sv[threadIdx.x + s];
            int   oi = si[threadIdx.x + s];
            if (ov > sv[threadIdx.x] ||
                (ov == sv[threadIdx.x] && oi < si[threadIdx.x])) {
                sv[threadIdx.x] = ov;
                si[threadIdx.x] = oi;
            }
        }
        __syncthreads();
    }
    if (threadIdx.x == 0) ntok[b] = si[0];
}

// ---------------------------------------------------------------------------
// Host launcher
// ---------------------------------------------------------------------------
static inline size_t alignup(size_t x) { return (x + 255) & ~(size_t)255; }

extern "C" void kernel_launch(void* const* d_in, const int* in_sizes, int n_in,
                              void* d_out, int out_size, void* d_ws, size_t ws_size,
                              hipStream_t stream) {
    (void)in_sizes; (void)n_in; (void)out_size; (void)ws_size;
    // inputs (setup_inputs dict order, params flattened in definition order)
    const float* hidden     = (const float*)d_in[0];   // [2,16,512]
    const int*   loc        = (const int*)d_in[1];
    const int*   gender     = (const int*)d_in[2];
    const int*   tags_token = (const int*)d_in[3];     // [16,8,8]
    const int*   tags_len   = (const int*)d_in[4];     // [16,8]
    const float* attn_value = (const float*)d_in[5];   // [16,64,512]
    const int*   attn_mask  = (const int*)d_in[6];     // [16,64]
    /* d_in[7] = num_steps (device scalar) -- STEPS_=32 is compile-time */
    const float* text_emb   = (const float*)d_in[8];   // [V,E]
    const float* loc_emb    = (const float*)d_in[9];
    const float* gen_emb    = (const float*)d_in[10];
    const float* W_dec      = (const float*)d_in[11];
    const float* W_tags     = (const float*)d_in[12];
    const float* W_prof     = (const float*)d_in[13];
    const float* r0_Wih     = (const float*)d_in[14];  // [1536,1024]
    const float* r0_Whh     = (const float*)d_in[15];  // [1536,512]
    const float* r0_bih     = (const float*)d_in[16];
    const float* r0_bhh     = (const float*)d_in[17];
    const float* r1_Wih     = (const float*)d_in[18];  // [1536,512]
    const float* r1_Whh     = (const float*)d_in[19];
    const float* r1_bih     = (const float*)d_in[20];
    const float* r1_bhh     = (const float*)d_in[21];
    const float* tr_Wih     = (const float*)d_in[22];  // [1536,512]
    const float* tr_Whh     = (const float*)d_in[23];
    const float* tr_bih     = (const float*)d_in[24];
    const float* tr_bhh     = (const float*)d_in[25];
    const float* Wp         = (const float*)d_in[26];  // [V,E]
    const float* Ws         = (const float*)d_in[27];  // [V,H]
    const float* bias_o     = (const float*)d_in[28];  // [1,V]
    const float* w_o        = (const float*)d_in[29];  // [1,H]

    float* out = (float*)d_out;                        // [B,STEPS,V]

    // workspace carve-up
    char* w = (char*)d_ws;
    const size_t VE = (size_t)V_ * E_;
    __bf16* Wp_bf   = (__bf16*)w;                 w += alignup(VE * 2);
    __bf16* Ws_bf   = (__bf16*)w;                 w += alignup(VE * 2);
    float*  hA      = (float*)w;                  w += alignup(2 * B_ * H_ * 4);
    float*  hB      = (float*)w;                  w += alignup(2 * B_ * H_ * 4);
    float*  thA     = (float*)w;                  w += alignup(BT_ * E_ * 4);
    float*  thB     = (float*)w;                  w += alignup(BT_ * E_ * 4);
    float*  hs_all  = (float*)w;                  w += alignup((size_t)TAGLEN_ * BT_ * E_ * 4);
    float*  xt      = (float*)w;                  w += alignup(BT_ * E_ * 4);
    float*  enc_tags= (float*)w;                  w += alignup(B_ * NTAGS_ * E_ * 4);
    float*  loc_e   = (float*)w;                  w += alignup(B_ * E_ * 4);
    float*  gen_e   = (float*)w;                  w += alignup(B_ * E_ * 4);
    float*  xbuf    = (float*)w;                  w += alignup(B_ * (E_ + DV_) * 4);
    __bf16* wp_bf   = (__bf16*)w;                 w += alignup(B_ * E_ * 2);
    __bf16* state_bf= (__bf16*)w;                 w += alignup(B_ * E_ * 2);
    float*  abuf    = (float*)w;                  w += alignup(B_ * 4);
    int*    ntok    = (int*)w;                    w += alignup(B_ * 4);

    // ---- one-time setup ----
    cvt_bf16_kernel<<<2048, 256, 0, stream>>>(Wp, Wp_bf, (int)VE);
    cvt_bf16_kernel<<<2048, 256, 0, stream>>>(Ws, Ws_bf, (int)VE);
    init_kernel<<<(BT_ * E_ + 255) / 256, 256, 0, stream>>>(hidden, hA, thA, ntok);
    embed_lg_kernel<<<(B_ * E_ + 255) / 256, 256, 0, stream>>>(
        loc, gender, loc_emb, gen_emb, loc_e, gen_e);

    // ---- tag encoder: GRU(E->E) over 128 sequences, 8 timesteps ----
    float* th_cur = thA;
    float* th_next = thB;
    for (int t = 0; t < TAGLEN_; ++t) {
        tag_gather_kernel<<<(BT_ * E_ + 255) / 256, 256, 0, stream>>>(
            text_emb, tags_token, t, xt);
        gru_layer_kernel<<<(BT_ * 512 + 255) / 256, 256, 0, stream>>>(
            xt, E_, th_cur, tr_Wih, tr_Whh, tr_bih, tr_bhh,
            th_next, nullptr, hs_all + (size_t)t * BT_ * E_, BT_);
        float* tmp = th_cur; th_cur = th_next; th_next = tmp;
    }
    tag_select_kernel<<<(BT_ * E_ + 255) / 256, 256, 0, stream>>>(
        hs_all, tags_len, enc_tags);

    // ---- autoregressive decode: 32 serial steps ----
    float* h_cur = hA;
    float* h_next = hB;
    const int NTILES = V_ / 16;                       // 3125
    const int LBLOCKS = (NTILES + 7) / 8;             // 8 waves/block
    for (int step = 0; step < STEPS_; ++step) {
        attend_kernel<<<B_, 256, 0, stream>>>(
            h_cur + B_ * H_, ntok, text_emb,
            W_dec, attn_value, attn_mask,
            W_tags, enc_tags, tags_len,
            W_prof, loc_e, gen_e,
            xbuf, wp_bf);
        // layer 0: x=[emb|ctx] (16x1024)
        gru_layer_kernel<<<(B_ * 512 + 255) / 256, 256, 0, stream>>>(
            xbuf, E_ + DV_, h_cur, r0_Wih, r0_Whh, r0_bih, r0_bhh,
            h_next, nullptr, nullptr, B_);
        // layer 1: x = new h0 (16x512); also emits state in bf16 for WMMA
        gru_layer_kernel<<<(B_ * 512 + 255) / 256, 256, 0, stream>>>(
            h_next, H_, h_cur + B_ * H_, r1_Wih, r1_Whh, r1_bih, r1_bhh,
            h_next + B_ * H_, state_bf, nullptr, B_);
        gate_kernel<<<1, 512, 0, stream>>>(h_next + B_ * H_, w_o, abuf);
        logits_wmma_kernel<<<LBLOCKS, 256, 0, stream>>>(
            wp_bf, state_bf, Wp_bf, Ws_bf, abuf, bias_o, out, step);
        argmax_kernel<<<B_, 256, 0, stream>>>(out, step, ntok);
        float* tmp = h_cur; h_cur = h_next; h_next = tmp;
    }
}
